// GCNModel_73478300500078
// MI455X (gfx1250) — matrix operand order repaired and checked by reading, not compiled, not run on hardware
//
#include <hip/hip_runtime.h>
#include <hip/hip_bf16.h>

// CDNA5 / gfx1250: wave32, WMMA 16x16x32 bf16 with f32 accumulate.
typedef __attribute__((ext_vector_type(16))) __bf16 v16bf;
typedef __attribute__((ext_vector_type(8)))  __bf16 v8bf;
typedef __attribute__((ext_vector_type(8)))  float  v8f;

// Types matching the async-LDS builtin's parameters (from hipcc diagnostics):
// param0: int __vector(4) __device__* (AS1), param1: AS3 variant.
typedef int v4i_vs __attribute__((vector_size(16)));
typedef __attribute__((address_space(1))) v4i_vs* g_v4i_p;
typedef __attribute__((address_space(3))) v4i_vs* l_v4i_p;

#if defined(__has_builtin)
#if __has_builtin(__builtin_amdgcn_global_load_async_to_lds_b128)
#define USE_ASYNC_LDS 1
#endif
#endif

// ---------------------------------------------------------------------------
// One-time weight prep: Wt[n][k] = bf16(W[k][n]), zero-padded to NP x Kpad.
// ---------------------------------------------------------------------------
template<int NCOLS, int NP>
__global__ void k_prep_w(const float* __restrict__ W, __bf16* __restrict__ Wt,
                         int K, int Kpad)
{
    int idx = blockIdx.x * 256 + threadIdx.x;
    if (idx >= NP * Kpad) return;
    int n = idx / Kpad, k = idx % Kpad;
    float v = (n < NCOLS && k < K) ? W[k * NCOLS + n] : 0.f;
    Wt[idx] = (__bf16)v;
}

// ---------------------------------------------------------------------------
// GEMM: C[M x NCOLS] = A[M x K] * Wt^T, fp32 A/C, bf16 WMMA inside.
// Block = 256 threads (8 waves). Block tile 128 x NP; wave tile 16 x NP.
// B tile staged from pre-transposed bf16 Wt via async global->LDS (CDNA5).
// K loop split into guard-free body + one guarded tail iteration.
// ---------------------------------------------------------------------------
template<int NCOLS, int NTILES>
__global__ __launch_bounds__(256)
void gemm_wmma_bf16(const float* __restrict__ A, const __bf16* __restrict__ Wt,
                    float* __restrict__ C, int M, int K, int Kpad)
{
    constexpr int NP = NTILES * 16;                 // padded N
    __shared__ __align__(64) __bf16 sA[128 * 32];   // [m][k]
    __shared__ __align__(64) __bf16 sB[NP * 32];    // [n][k] (K-transposed)

    const int tid    = threadIdx.x;
    const int lane   = tid & 31;
    const int wave   = tid >> 5;
    const int blockM = blockIdx.x * 128;

    const v8f vzero = {0.f, 0.f, 0.f, 0.f, 0.f, 0.f, 0.f, 0.f};
    v8f acc[NTILES];
#pragma unroll
    for (int t = 0; t < NTILES; ++t) acc[t] = vzero;

    // A staging: thread -> (row, 16-elem half). Row clamped to valid memory;
    // out-of-range rows only feed out-of-range C rows, masked at store.
    const int sRow  = tid >> 1;             // 0..127
    const int sHalf = (tid & 1) * 16;       // 0 or 16
    const int gRow  = blockM + sRow;
    const int cRow  = (gRow < M) ? gRow : (M - 1);
    const float* arow = A + (long long)cRow * K;

    // Fragment addressing (wave32), contiguous k = 16*(lane>>4) + s for both A and B:
    const int aRow = wave * 16 + (lane & 15);
    const int kbA  = (lane >> 4) * 16;
    const int nLn  = lane & 15;
    const int kbB  = (lane >> 4) * 16;

    auto stageB = [&](int k0) {
        constexpr int BUNITS = NP * 4;              // NP*32*2B / 16B
#pragma unroll 2
        for (int i = tid; i < BUNITS; i += 256) {
            int n = i >> 2, c = (i & 3) * 8;
            const __bf16* src = Wt + (long long)n * Kpad + k0 + c;
            __bf16*       dst = &sB[n * 32 + c];
#ifdef USE_ASYNC_LDS
            __builtin_amdgcn_global_load_async_to_lds_b128(
                (g_v4i_p)(__bf16*)src, (l_v4i_p)dst, 0, 0);
#else
            *(v8bf*)dst = *(const v8bf*)src;
#endif
        }
#ifdef USE_ASYNC_LDS
        asm volatile("s_wait_asynccnt 0x0" ::: "memory");
#endif
    };

    auto compute = [&]() {
        const __bf16* ap = &sA[aRow * 32];
        v8bf alo = *(const v8bf*)(ap + kbA);        // K = Kb .. Kb+7
        v8bf ahi = *(const v8bf*)(ap + kbA + 8);    // K = Kb+8 .. Kb+15
        v16bf afrag = __builtin_shufflevector(alo, ahi,
            0, 1, 2, 3, 4, 5, 6, 7, 8, 9, 10, 11, 12, 13, 14, 15);
#pragma unroll
        for (int t = 0; t < NTILES; ++t) {
            v16bf bfrag = *(const v16bf*)(&sB[(t * 16 + nLn) * 32 + kbB]);
            acc[t] = __builtin_amdgcn_wmma_f32_16x16x32_bf16(
                false, afrag, false, bfrag, (short)0, acc[t], false, false);
        }
    };

    const int kBody = K & ~31;                      // full 32-wide tiles

    // ---- Guard-free body ----
    for (int k0 = 0; k0 < kBody; k0 += 32) {
        v16bf t;
#pragma unroll
        for (int j = 0; j < 4; ++j) {
            float4 f = *(const float4*)(arow + k0 + sHalf + j * 4);
            t[j * 4 + 0] = (__bf16)f.x;
            t[j * 4 + 1] = (__bf16)f.y;
            t[j * 4 + 2] = (__bf16)f.z;
            t[j * 4 + 3] = (__bf16)f.w;
        }
        *(v16bf*)(&sA[sRow * 32 + sHalf]) = t;
        stageB(k0);
        __syncthreads();
        compute();
        __syncthreads();
    }

    // ---- Guarded K tail (zero-fill; avoids NaN garbage x 0 pollution) ----
    if (kBody < K) {
        v16bf t;
#pragma unroll
        for (int j = 0; j < 16; ++j) {
            int gk = kBody + sHalf + j;
            t[j] = (gk < K) ? (__bf16)arow[gk] : (__bf16)0.f;
        }
        *(v16bf*)(&sA[sRow * 32 + sHalf]) = t;
        stageB(kBody);                              // Wt zero-padded to Kpad
        __syncthreads();
        compute();
        __syncthreads();
    }

    // C/D layout: VGPR r -> M = 8*(lane>=16) + r ; N = lane&15.
    const int rowBase = blockM + wave * 16 + ((lane >> 4) << 3);
#pragma unroll
    for (int t = 0; t < NTILES; ++t) {
        int col = t * 16 + nLn;
#pragma unroll
        for (int r = 0; r < 8; ++r) {
            int row = rowBase + r;
            if (row < M && col < NCOLS)
                C[(long long)row * NCOLS + col] = acc[t][r];
        }
    }
}

// ---------------------------------------------------------------------------
// Graph kernels
// ---------------------------------------------------------------------------
__global__ void k_deg_init(float* deg, int n) {
    int i = blockIdx.x * 256 + threadIdx.x;
    if (i < n) deg[i] = 1.0f;                 // self-loop contributes 1
}

__global__ void k_deg_accum(const int* __restrict__ dst, float* deg, int E) {
    int i = blockIdx.x * 256 + threadIdx.x;
    if (i < E) atomicAdd(&deg[dst[i]], 1.0f);
}

__global__ void k_rsqrt(float* d, int n) {
    int i = blockIdx.x * 256 + threadIdx.x;
    if (i < n) d[i] = rsqrtf(d[i]);           // deg >= 1 always
}

// msg = h[src] * dinv[src]*dinv[dst]; scatter-add into agg[dst].
// Each thread handles 4 features of one edge (float4 gather, 4 f32 atomics).
template<int F>
__global__ void k_edge_agg(const int* __restrict__ src, const int* __restrict__ dst,
                           const float* __restrict__ dinv, const float* __restrict__ h,
                           float* __restrict__ agg, long long total4)
{
    constexpr int FQ = F / 4;
    long long idx = (long long)blockIdx.x * blockDim.x + threadIdx.x;
    if (idx >= total4) return;
    int e = (int)(idx / FQ);
    int f = (int)(idx % FQ) * 4;
    int s = src[e], d = dst[e];
    float w = dinv[s] * dinv[d];
    float4 hv = *(const float4*)(h + (long long)s * F + f);
    float* ap = agg + (long long)d * F + f;
    atomicAdd(ap + 0, hv.x * w);
    atomicAdd(ap + 1, hv.y * w);
    atomicAdd(ap + 2, hv.z * w);
    atomicAdd(ap + 3, hv.w * w);
}

// inout = inout + h*dinv[i]^2 + bias  [self-loop + bias], optional ReLU.
template<int F, bool RELU>
__global__ void k_finish(const float* __restrict__ h, const float* __restrict__ dinv,
                         const float* __restrict__ bias, float* inout, int n)
{
    constexpr int FQ = F / 4;
    long long idx = (long long)blockIdx.x * blockDim.x + threadIdx.x;
    if (idx >= (long long)n * FQ) return;
    int i = (int)(idx / FQ);
    int f = (int)(idx % FQ) * 4;
    float dd = dinv[i] * dinv[i];
    long long base = (long long)i * F + f;
    float4 hv = *(const float4*)(h + base);
    float4 av = *(float4*)(inout + base);
    float4 r;
    r.x = av.x + hv.x * dd + bias[f + 0];
    r.y = av.y + hv.y * dd + bias[f + 1];
    r.z = av.z + hv.z * dd + bias[f + 2];
    r.w = av.w + hv.w * dd + bias[f + 3];
    if (RELU) {
        r.x = fmaxf(r.x, 0.f); r.y = fmaxf(r.y, 0.f);
        r.z = fmaxf(r.z, 0.f); r.w = fmaxf(r.w, 0.f);
    }
    *(float4*)(inout + base) = r;
}

// In-place log_softmax over 40 columns; one wave32 per row (cols lane, lane+32).
__global__ void k_log_softmax40(float* __restrict__ io, int n) {
    int lane = threadIdx.x & 31;
    int warp = threadIdx.x >> 5;
    int row  = blockIdx.x * 8 + warp;
    if (row >= n) return;
    float* p = io + (long long)row * 40;
    float v0 = p[lane];
    bool  hi = (lane + 32) < 40;
    float v1 = hi ? p[lane + 32] : -__builtin_inff();
    float m = fmaxf(v0, v1);
    for (int off = 16; off > 0; off >>= 1) m = fmaxf(m, __shfl_xor(m, off, 32));
    float s = __expf(v0 - m) + (hi ? __expf(v1 - m) : 0.0f);
    for (int off = 16; off > 0; off >>= 1) s += __shfl_xor(s, off, 32);
    float lse = m + __logf(s);
    p[lane] = v0 - lse;
    if (hi) p[lane + 32] = v1 - lse;
}

// ---------------------------------------------------------------------------
// Launch
// ---------------------------------------------------------------------------
extern "C" void kernel_launch(void* const* d_in, const int* in_sizes, int n_in,
                              void* d_out, int out_size, void* d_ws, size_t ws_size,
                              hipStream_t stream)
{
    const float* x  = (const float*)d_in[0];
    const int*   ei = (const int*)d_in[1];   // [2 x E], row0 = src, row1 = dst
    const float* W1 = (const float*)d_in[2];
    const float* b1 = (const float*)d_in[3];
    const float* W2 = (const float*)d_in[4];
    const float* b2 = (const float*)d_in[5];

    const int E   = in_sizes[1] / 2;
    const int HID = in_sizes[3];            // 128
    const int OUT = in_sizes[5];            // 40
    const int IN  = in_sizes[2] / HID;      // 500
    const int N   = in_sizes[0] / IN;       // 100000

    const int Kpad1 = (IN  + 31) & ~31;     // 512
    const int Kpad2 = (HID + 31) & ~31;     // 128

    const int* src = ei;
    const int* dst = ei + E;

    // Workspace: dinv[N] | h1[N*128] | g1[N*128] | Wt1 bf16[128*Kpad1] | Wt2 bf16[48*Kpad2]
    char* wsb = (char*)d_ws;
    size_t o = 0;
    float* dinv = (float*)(wsb + o); o = (o + (size_t)N * 4 + 255) & ~(size_t)255;
    float* h1   = (float*)(wsb + o); o += (size_t)N * 128 * 4;
    float* g1   = (float*)(wsb + o); o += (size_t)N * 128 * 4;
    __bf16* Wt1 = (__bf16*)(wsb + o); o = (o + (size_t)128 * Kpad1 * 2 + 255) & ~(size_t)255;
    __bf16* Wt2 = (__bf16*)(wsb + o);
    float* h2   = h1;                       // h1 dead after layer-1 finish
    float* out  = (float*)d_out;

    // Zero accumulation targets (capture-safe async memsets).
    (void)hipMemsetAsync(g1,  0, (size_t)N * HID * sizeof(float), stream);
    (void)hipMemsetAsync(out, 0, (size_t)N * OUT * sizeof(float), stream);

    // Degrees -> dinv ; transpose/convert weights once.
    k_deg_init <<<(N + 255) / 256, 256, 0, stream>>>(dinv, N);
    k_deg_accum<<<(E + 255) / 256, 256, 0, stream>>>(dst, dinv, E);
    k_rsqrt    <<<(N + 255) / 256, 256, 0, stream>>>(dinv, N);
    k_prep_w<128, 128><<<(128 * Kpad1 + 255) / 256, 256, 0, stream>>>(W1, Wt1, IN,  Kpad1);
    k_prep_w<40,  48> <<<(48  * Kpad2 + 255) / 256, 256, 0, stream>>>(W2, Wt2, HID, Kpad2);

    // Layer 1: h1 = x @ W1 (WMMA), aggregate, +self-loop +b1, ReLU.
    gemm_wmma_bf16<128, 8><<<(N + 127) / 128, 256, 0, stream>>>(x, Wt1, h1, N, IN, Kpad1);
    {
        long long tot4 = (long long)E * (HID / 4);
        k_edge_agg<128><<<(unsigned)((tot4 + 255) / 256), 256, 0, stream>>>(src, dst, dinv, h1, g1, tot4);
        long long nel4 = (long long)N * (HID / 4);
        k_finish<128, true><<<(unsigned)((nel4 + 255) / 256), 256, 0, stream>>>(h1, dinv, b1, g1, N);
    }

    // Layer 2: h2 = g1 @ W2 (WMMA), aggregate into d_out, +self-loop +b2.
    gemm_wmma_bf16<40, 3><<<(N + 127) / 128, 256, 0, stream>>>(g1, Wt2, h2, N, HID, Kpad2);
    {
        long long tot4 = (long long)E * (OUT / 4);
        k_edge_agg<40><<<(unsigned)((tot4 + 255) / 256), 256, 0, stream>>>(src, dst, dinv, h2, out, tot4);
        long long nel4 = (long long)N * (OUT / 4);
        k_finish<40, false><<<(unsigned)((nel4 + 255) / 256), 256, 0, stream>>>(h2, dinv, b2, out, N);
    }

    // log_softmax rows of 40, in place.
    k_log_softmax40<<<(N + 7) / 8, 256, 0, stream>>>(out, N);
}
